// SAT4Crystals_11338713661566
// MI455X (gfx1250) — matrix operand order repaired and checked by reading, not compiled, not run on hardware
//
#include <hip/hip_runtime.h>
#include <hip/hip_bf16.h>
#include <math.h>

#define HID 64

typedef __attribute__((ext_vector_type(16))) __bf16 v16bf;
typedef __attribute__((ext_vector_type(8)))  float  v8f;

union Frag { unsigned u[8]; v16bf h; };

__device__ __forceinline__ unsigned short f2b(float f) {
  unsigned u = __float_as_uint(f);
  u += 0x7FFFu + ((u >> 16) & 1u);          // round-to-nearest-even to bf16
  return (unsigned short)(u >> 16);
}
__device__ __forceinline__ unsigned pack2(float a, float b) {
  return (unsigned)f2b(a) | ((unsigned)f2b(b) << 16);
}
// order-preserving float <-> uint encoding for atomicMax on floats
__device__ __forceinline__ unsigned fenc(float f) {
  unsigned u = __float_as_uint(f);
  return (u >> 31) ? ~u : (u | 0x80000000u);
}
__device__ __forceinline__ float fdec(unsigned u) {
  return __uint_as_float((u >> 31) ? (u & 0x7FFFFFFFu) : ~u);
}

// ---------------------------------------------------------------------------
// WMMA GEMM: C[M x Nc] = A[M x K] (f32, cvt->bf16 inline) * B (bf16, stored
// transposed as BT[Nc x K]) + bias.  Block = 8 waves; wave computes a 16x64
// strip (4 N-tiles), block covers 128 rows.  K, Nc multiples of 32 / 64.
// ---------------------------------------------------------------------------
__global__ void wmma_gemm(const float* __restrict__ A,
                          const unsigned short* __restrict__ BT,
                          const float* __restrict__ bias,
                          float* __restrict__ C, int M, int K, int Nc) {
  const int lane  = threadIdx.x & 31;
  const int wave  = threadIdx.x >> 5;
  const int half  = lane >> 4;     // 0 | 1
  const int lm    = lane & 15;
  const int tileM = blockIdx.x * 128 + wave * 16;
  const int baseN = blockIdx.y * 64;

  int m = tileM + lm;
  if (m > M - 1) m = M - 1;        // branchless clamp keeps EXEC all-1s
  const float* arow = A + (size_t)m * K;

  v8f acc[4] = {};
  for (int kk = 0; kk < K; kk += 32) {
    // A fragment: 16x32 bf16.  lanes<16: e<8 -> K=e, e>=8 -> K=e+8; lanes>=16: +8
    Frag a;
    const float* ap = arow + kk + half * 8;
#pragma unroll
    for (int v = 0; v < 8; ++v) {
      int k0 = (v < 4) ? (2 * v) : (8 + 2 * v);   // v>=4: 16+2(v-4)
      a.u[v] = pack2(ap[k0], ap[k0 + 1]);
    }
#pragma unroll
    for (int t = 0; t < 4; ++t) {
      // B fragment: 32x16.  n = lane&15, K same as A: e<8 -> e, e>=8 -> e+8; lanes>=16: +8
      Frag b;
      const unsigned short* bp =
          BT + (size_t)(baseN + t * 16 + lm) * K + kk + half * 8;
#pragma unroll
      for (int v = 0; v < 8; ++v) b.u[v] = *(const unsigned*)(bp + ((v < 4) ? (2 * v) : (8 + 2 * v)));
      acc[t] = __builtin_amdgcn_wmma_f32_16x16x32_bf16(
          false, a.h, false, b.h, (short)0, acc[t], false, false);
    }
  }
  // C/D layout: n = lane&15, m = r + (lane>=16 ? 8 : 0)
  const int mr0 = tileM + half * 8;
#pragma unroll
  for (int t = 0; t < 4; ++t) {
    int n = baseN + t * 16 + lm;
    float bv = bias ? bias[n] : 0.0f;
#pragma unroll
    for (int r = 0; r < 8; ++r) {
      int mr = mr0 + r;
      if (mr < M) C[(size_t)mr * Nc + n] = acc[t][r] + bv;
    }
  }
}

// ---------------------------------------------------------------------------
// small helpers
// ---------------------------------------------------------------------------
__global__ void stage_wt(const float* __restrict__ W, unsigned short* __restrict__ WT,
                         int K, int Nc) {   // W[K x Nc] f32 -> WT[Nc x K] bf16
  int i = blockIdx.x * blockDim.x + threadIdx.x;
  if (i >= K * Nc) return;
  int k = i / Nc, n = i - k * Nc;
  WT[(size_t)n * K + k] = f2b(W[i]);
}

__global__ void fill_u32(unsigned* __restrict__ p, unsigned v, size_t n) {
  size_t i = (size_t)blockIdx.x * blockDim.x + threadIdx.x;
  if (i < n) p[i] = v;
}

__global__ void embed_k(const float* __restrict__ x, const float* __restrict__ W,
                        const float* __restrict__ b, float* __restrict__ out,
                        int rows, int D) {  // out = x[rows x D] @ W[D x 64] + b
  size_t i = (size_t)blockIdx.x * blockDim.x + threadIdx.x;
  if (i >= (size_t)rows * HID) return;
  int r = (int)(i >> 6), c = (int)(i & 63);
  float acc = b[c];
  for (int d = 0; d < D; ++d) acc += x[(size_t)r * D + d] * W[d * HID + c];
  out[i] = acc;
}

__global__ void gather64(const float* __restrict__ src, const int* __restrict__ idx,
                         float* __restrict__ out, int rows) {
  size_t i = (size_t)blockIdx.x * blockDim.x + threadIdx.x;
  if (i >= (size_t)rows * HID) return;
  int r = (int)(i >> 6), c = (int)(i & 63);
  out[i] = src[(size_t)idx[r] * HID + c];
}

__global__ void copy_off(const float* __restrict__ in, float* __restrict__ out,
                         int rows, int ostride, int ooff) {
  size_t i = (size_t)blockIdx.x * blockDim.x + threadIdx.x;
  if (i >= (size_t)rows * HID) return;
  int r = (int)(i >> 6), c = (int)(i & 63);
  out[(size_t)r * ostride + ooff + c] = in[i];
}

__global__ void addk(const float* __restrict__ a, const float* __restrict__ b,
                     float* __restrict__ out, size_t n) {
  size_t i = (size_t)blockIdx.x * blockDim.x + threadIdx.x;
  if (i < n) out[i] = a[i] + b[i];
}

__global__ void seg_accum(const float* __restrict__ x, const int* __restrict__ idx,
                          float* __restrict__ sum, float* __restrict__ cnt, int rows) {
  size_t i = (size_t)blockIdx.x * blockDim.x + threadIdx.x;
  if (i >= (size_t)rows * HID) return;
  int r = (int)(i >> 6), c = (int)(i & 63);
  int t = idx[r];
  atomicAdd(sum + (size_t)t * HID + c, x[i]);
  if (c == 0) atomicAdd(cnt + t, 1.0f);
}

__global__ void seg_div_store(const float* __restrict__ sum, const float* __restrict__ cnt,
                              float* __restrict__ out, int segs, int ostride, int ooff) {
  size_t i = (size_t)blockIdx.x * blockDim.x + threadIdx.x;
  if (i >= (size_t)segs * HID) return;
  int r = (int)(i >> 6), c = (int)(i & 63);
  out[(size_t)r * ostride + ooff + c] = sum[i] / fmaxf(cnt[r], 1.0f);
}

// ---------------------------------------------------------------------------
// attention passes (segment softmax over dst)
// ---------------------------------------------------------------------------
__global__ void attn_alpha(const float* __restrict__ qkv, const float* __restrict__ ef,
                           const int* __restrict__ src, const int* __restrict__ dst,
                           float* __restrict__ alpha, unsigned* __restrict__ amax,
                           int Ee, int H, float scale) {
  size_t i = (size_t)blockIdx.x * blockDim.x + threadIdx.x;
  if (i >= (size_t)Ee * H) return;
  int ee = (int)(i / H), hh = (int)(i % H);
  int Cc = HID / H;
  int sn = src[ee], dn = dst[ee];
  const float* q = qkv + (size_t)dn * 192 + hh * Cc;
  const float* k = qkv + (size_t)sn * 192 + 64 + hh * Cc;
  const float* e = ef  + (size_t)ee * HID + hh * Cc;
  float acc = 0.f;
  for (int c = 0; c < Cc; ++c) acc += q[c] * (k[c] + e[c]);
  acc *= scale;
  alpha[i] = acc;
  atomicMax(amax + (size_t)dn * H + hh, fenc(acc));
}

__global__ void amax_fix(unsigned* __restrict__ am, int n) {  // decode in place
  int i = blockIdx.x * blockDim.x + threadIdx.x;
  if (i >= n) return;
  float f = fdec(am[i]);
  if (!__builtin_isfinite(f)) f = 0.f;   // empty segment -> 0 (matches reference)
  am[i] = __float_as_uint(f);
}

__global__ void attn_pass2(float* __restrict__ alpha, const float* __restrict__ amaxf,
                           float* __restrict__ denom, const int* __restrict__ dst,
                           int Ee, int H) {
  size_t i = (size_t)blockIdx.x * blockDim.x + threadIdx.x;
  if (i >= (size_t)Ee * H) return;
  int ee = (int)(i / H), hh = (int)(i % H);
  int dn = dst[ee];
  float p = __expf(alpha[i] - amaxf[(size_t)dn * H + hh]);
  alpha[i] = p;
  atomicAdd(denom + (size_t)dn * H + hh, p);
}

__global__ void attn_agg(const float* __restrict__ qkv, const float* __restrict__ ef,
                         const float* __restrict__ p, const float* __restrict__ denom,
                         const int* __restrict__ src, const int* __restrict__ dst,
                         float* __restrict__ out, int Ee, int H) {
  size_t i = (size_t)blockIdx.x * blockDim.x + threadIdx.x;
  if (i >= (size_t)Ee * HID) return;
  int ee = (int)(i >> 6), c = (int)(i & 63);
  int Cc = HID / H, hh = c / Cc;
  int sn = src[ee], dn = dst[ee];
  float a = p[(size_t)ee * H + hh] / (denom[(size_t)dn * H + hh] + 1e-16f);
  float val = (qkv[(size_t)sn * 192 + 128 + c] + ef[(size_t)ee * HID + c]) * a;
  atomicAdd(out + (size_t)dn * HID + c, val);
}

// ---------------------------------------------------------------------------
// final MLP head: out[g] = relu(g @ W1 + b1) @ W2 + b2
// ---------------------------------------------------------------------------
__global__ void mlp_head(const float* __restrict__ g, const float* __restrict__ W1,
                         const float* __restrict__ b1, const float* __restrict__ W2,
                         const float* __restrict__ b2, float* __restrict__ out) {
  int gg = blockIdx.x, j = threadIdx.x;   // 32 threads
  __shared__ float hid[32];
  float acc = b1[j];
  for (int i = 0; i < HID; ++i) acc += g[(size_t)gg * HID + i] * W1[i * 32 + j];
  hid[j] = fmaxf(acc, 0.f);
  __syncthreads();
  if (j == 0) {
    float o = b2[0];
    for (int t = 0; t < 32; ++t) o += hid[t] * W2[t];
    out[gg] = o;
  }
}

// ---------------------------------------------------------------------------
extern "C" void kernel_launch(void* const* d_in, const int* in_sizes, int n_in,
                              void* d_out, int out_size, void* d_ws, size_t ws_size,
                              hipStream_t stream) {
  (void)n_in; (void)ws_size;
  const float* x         = (const float*)d_in[0];
  const float* edge_attr = (const float*)d_in[1];
  const int*   eidx      = (const int*)d_in[2];
  const int*   seidx     = (const int*)d_in[3];
  const int*   sind      = (const int*)d_in[4];
  const int*   batch     = (const int*)d_in[5];
  const float* W_node = (const float*)d_in[6],  *b_node = (const float*)d_in[7];
  const float* W_edge = (const float*)d_in[8],  *b_edge = (const float*)d_in[9];
  const float* Wqkv1  = (const float*)d_in[10], *bqkv1  = (const float*)d_in[11];
  const float* We1    = (const float*)d_in[12], *Ws1    = (const float*)d_in[13];
  const float* bs1    = (const float*)d_in[14];
  const float* Wqkv2  = (const float*)d_in[15], *bqkv2  = (const float*)d_in[16];
  const float* We2    = (const float*)d_in[17], *Ws2    = (const float*)d_in[18];
  const float* bs2    = (const float*)d_in[19];
  const float* Wqkva  = (const float*)d_in[20], *bqkva  = (const float*)d_in[21];
  const float* Wea    = (const float*)d_in[22], *Wsa    = (const float*)d_in[23];
  const float* bsa    = (const float*)d_in[24];
  const float* W_h1   = (const float*)d_in[25], *b_h1   = (const float*)d_in[26];
  const float* W_h2   = (const float*)d_in[27], *b_h2   = (const float*)d_in[28];

  const int N = in_sizes[4];
  const int E = in_sizes[2] / 2;
  const int G = out_size;
  const int L = in_sizes[10] / (HID * 3 * HID);
  const int *src = eidx, *dst = eidx + E;
  const int *ssrc = seidx, *sdst = seidx + N;

  char* w = (char*)d_ws;
  size_t off = 0;
  auto alloc = [&](size_t bytes) -> void* {
    off = (off + 255) & ~(size_t)255; void* p = w + off; off += bytes; return p;
  };
  float* h     = (float*)alloc((size_t)N * 64 * 4);
  float* s     = (float*)alloc((size_t)N * 64 * 4);
  float* tout  = (float*)alloc((size_t)N * 64 * 4);
  float* skip  = (float*)alloc((size_t)N * 64 * 4);
  float* hcat  = (float*)alloc((size_t)N * 128 * 4);
  float* qkv   = (float*)alloc((size_t)N * 192 * 4);
  float* subse = (float*)alloc((size_t)N * 64 * 4);
  float* subef = (float*)alloc((size_t)N * 64 * 4);
  float* ssum  = (float*)alloc((size_t)N * 64 * 4);
  float* cnt   = (float*)alloc((size_t)N * 4);
  float* gbuf  = (float*)alloc((size_t)G * 64 * 4);
  unsigned* amax = (unsigned*)alloc((size_t)N * 8 * 4);
  float* denom = (float*)alloc((size_t)N * 8 * 4);
  float* alp   = (float*)alloc((size_t)E * 8 * 4);
  float* ea    = (float*)alloc((size_t)E * 64 * 4);
  float* eaf   = (float*)alloc((size_t)E * 64 * 4);
  unsigned short* wA = (unsigned short*)alloc((size_t)192 * 128 * 2);
  unsigned short* wB = (unsigned short*)alloc((size_t)128 * 64 * 2);

  auto cd = [](size_t a, size_t b) { return (unsigned)((a + b - 1) / b); };
  auto gemm = [&](const float* A, const unsigned short* BT, const float* bias,
                  float* C, int M, int K, int Nc) {
    dim3 grid(cd(M, 128), Nc / 64);
    wmma_gemm<<<grid, 256, 0, stream>>>(A, BT, bias, C, M, K, Nc);
  };
  auto tconv = [&](const float* xin, int K, int H, const int* es, const int* ed,
                   int Ee, const float* ef, const float* Wqkv, const float* bqkv,
                   const float* Ws, const float* bs, float* outp) {
    stage_wt<<<cd((size_t)K * 192, 256), 256, 0, stream>>>(Wqkv, wA, K, 192);
    gemm(xin, wA, bqkv, qkv, N, K, 192);
    fill_u32<<<cd((size_t)N * H, 256), 256, 0, stream>>>(amax, 0x007FFFFFu, (size_t)N * H);
    fill_u32<<<cd((size_t)N * H, 256), 256, 0, stream>>>((unsigned*)denom, 0u, (size_t)N * H);
    fill_u32<<<cd((size_t)N * 64, 256), 256, 0, stream>>>((unsigned*)tout, 0u, (size_t)N * 64);
    float scale = 1.0f / sqrtf((float)(HID / H));
    attn_alpha<<<cd((size_t)Ee * H, 256), 256, 0, stream>>>(qkv, ef, es, ed, alp, amax, Ee, H, scale);
    amax_fix<<<cd((size_t)N * H, 256), 256, 0, stream>>>(amax, N * H);
    attn_pass2<<<cd((size_t)Ee * H, 256), 256, 0, stream>>>(alp, (const float*)amax, denom, ed, Ee, H);
    attn_agg<<<cd((size_t)Ee * 64, 256), 256, 0, stream>>>(qkv, ef, alp, denom, es, ed, tout, Ee, H);
    stage_wt<<<cd((size_t)K * 64, 256), 256, 0, stream>>>(Ws, wB, K, 64);
    gemm(xin, wB, bs, skip, N, K, 64);
    addk<<<cd((size_t)N * 64, 256), 256, 0, stream>>>(tout, skip, outp, (size_t)N * 64);
  };

  // initial embeddings + subgraph edge-feature gather
  embed_k<<<cd((size_t)N * 64, 256), 256, 0, stream>>>(x, W_node, b_node, h, N, 3);
  embed_k<<<cd((size_t)E * 64, 256), 256, 0, stream>>>(edge_attr, W_edge, b_edge, ea, E, 4);
  gather64<<<cd((size_t)N * 64, 256), 256, 0, stream>>>(ea, sind, subse, N);

  for (int l = 0; l < L; ++l) {
    stage_wt<<<cd(64 * 64, 256), 256, 0, stream>>>(We1 + (size_t)l * 64 * 64, wA, 64, 64);
    gemm(subse, wA, nullptr, subef, N, 64, 64);
    tconv(h, 64, 4, ssrc, sdst, N, subef, Wqkv1 + (size_t)l * 64 * 192,
          bqkv1 + l * 192, Ws1 + (size_t)l * 64 * 64, bs1 + l * 64, s);

    stage_wt<<<cd(64 * 64, 256), 256, 0, stream>>>(We2 + (size_t)l * 64 * 64, wA, 64, 64);
    gemm(subse, wA, nullptr, subef, N, 64, 64);
    tconv(s, 64, 4, ssrc, sdst, N, subef, Wqkv2 + (size_t)l * 64 * 192,
          bqkv2 + l * 192, Ws2 + (size_t)l * 64 * 64, bs2 + l * 64, s);

    // struct = seg_mean(s, sind, N); hcat = [h | struct]
    fill_u32<<<cd((size_t)N * 64, 256), 256, 0, stream>>>((unsigned*)ssum, 0u, (size_t)N * 64);
    fill_u32<<<cd((size_t)N, 256), 256, 0, stream>>>((unsigned*)cnt, 0u, (size_t)N);
    seg_accum<<<cd((size_t)N * 64, 256), 256, 0, stream>>>(s, sind, ssum, cnt, N);
    copy_off<<<cd((size_t)N * 64, 256), 256, 0, stream>>>(h, hcat, N, 128, 0);
    seg_div_store<<<cd((size_t)N * 64, 256), 256, 0, stream>>>(ssum, cnt, hcat, N, 128, 64);

    // eaf = ea @ Wea[l]  (E x 64 GEMM via WMMA)
    stage_wt<<<cd(64 * 64, 256), 256, 0, stream>>>(Wea + (size_t)l * 64 * 64, wA, 64, 64);
    gemm(ea, wA, nullptr, eaf, E, 64, 64);
    tconv(hcat, 128, 8, src, dst, E, eaf, Wqkva + (size_t)l * 128 * 192,
          bqkva + l * 192, Wsa + (size_t)l * 128 * 64, bsa + l * 64, h);
  }

  // global mean pool + head
  fill_u32<<<cd((size_t)G * 64, 256), 256, 0, stream>>>((unsigned*)ssum, 0u, (size_t)G * 64);
  fill_u32<<<cd((size_t)G, 256), 256, 0, stream>>>((unsigned*)cnt, 0u, (size_t)G);
  seg_accum<<<cd((size_t)N * 64, 256), 256, 0, stream>>>(h, batch, ssum, cnt, N);
  seg_div_store<<<cd((size_t)G * 64, 256), 256, 0, stream>>>(ssum, cnt, gbuf, G, 64, 0);
  mlp_head<<<G, 32, 0, stream>>>(gbuf, W_h1, b_h1, W_h2, b_h2, (float*)d_out);
}
